// TopKMLPSAE_44160853737879
// MI455X (gfx1250) — matrix operand order repaired and checked by reading, not compiled, not run on hardware
//
#include <hip/hip_runtime.h>
#include <hip/hip_bf16.h>

// Problem constants (fixed by the reference).
#define SB 4096    // batch B
#define SC 768     // C
#define SD 3072    // D
#define SN 32768   // N
#define SK 16      // top-k

typedef __attribute__((ext_vector_type(16))) __bf16 v16bf;
typedef __attribute__((ext_vector_type(8)))  __bf16 v8bf;
typedef __attribute__((ext_vector_type(8)))  float  v8f;

__device__ __forceinline__ unsigned short f2bf(float f) {
    unsigned int u = __float_as_uint(f);
    u += 0x7FFFu + ((u >> 16) & 1u);        // round-to-nearest-even
    return (unsigned short)(u >> 16);
}
__device__ __forceinline__ float bf2f(unsigned short h) {
    return __uint_as_float(((unsigned int)h) << 16);
}
__device__ __forceinline__ float gelu_erf(float x) {
    return 0.5f * x * (1.0f + erff(x * 0.70710678118654752f));
}

// ---------------------------------------------------------------------------
// Elementwise casts
// ---------------------------------------------------------------------------
__global__ void k_cvt_bf16(const float* __restrict__ in,
                           unsigned short* __restrict__ out, size_t n) {
    size_t i = (size_t)blockIdx.x * blockDim.x + threadIdx.x;
    if (i < n) out[i] = f2bf(in[i]);
}

// xs = bf16(x - dec_b2), row length = SC
__global__ void k_sub_bf16(const float* __restrict__ x, const float* __restrict__ b,
                           unsigned short* __restrict__ out, size_t n) {
    size_t i = (size_t)blockIdx.x * blockDim.x + threadIdx.x;
    if (i < n) out[i] = f2bf(x[i] - b[i % SC]);
}

// ---------------------------------------------------------------------------
// decoder_w1 [D,N] f32  ->  w1T [N,D] bf16   (tiled LDS transpose, 32x32 tiles)
// ---------------------------------------------------------------------------
__global__ __launch_bounds__(256) void k_transpose_bf16(
    const float* __restrict__ W, unsigned short* __restrict__ WT) {
    __shared__ float tile[32][33];
    const int nb = blockIdx.x * 32;          // N tile
    const int db = blockIdx.y * 32;          // D tile
    const int tx = threadIdx.x & 31;
    const int ty = threadIdx.x >> 5;         // 0..7
#pragma unroll
    for (int i = 0; i < 32; i += 8)
        tile[ty + i][tx] = W[(size_t)(db + ty + i) * SN + (nb + tx)];
    __syncthreads();
#pragma unroll
    for (int i = 0; i < 32; i += 8)
        WT[(size_t)(nb + ty + i) * SD + (db + tx)] = f2bf(tile[tx][ty + i]);
}

// ---------------------------------------------------------------------------
// WMMA bf16 GEMM:  Out[M,N] = A[M,K] * B[N,K]^T + bias[N]   (row-major, K%32==0)
// Block: 256 thr = 8 waves (4 in M x 2 in N); block tile 256x128.
// Wave tile 64x64 = 4x4 fragments of v_wmma_f32_16x16x32_bf16.
// EPI: 0 = f32 store, 1 = erf-GELU then bf16 store.
// ---------------------------------------------------------------------------
template <int EPI>
__global__ __launch_bounds__(256) void k_gemm_bf16(
    const unsigned short* __restrict__ A, const unsigned short* __restrict__ Bm,
    const float* __restrict__ bias, void* __restrict__ Out,
    int M, int N, int K) {
    const int lane = threadIdx.x & 31;
    const int wave = threadIdx.x >> 5;
    const int m0 = blockIdx.x * 256 + (wave >> 1) * 64;
    const int n0 = blockIdx.y * 128 + (wave & 1) * 64;
    const int lr = lane & 15;   // fragment row (A) / col (B/C)
    const int lh = lane >> 4;   // K-half selector

    v8f acc[4][4];
#pragma unroll
    for (int i = 0; i < 4; i++)
#pragma unroll
        for (int j = 0; j < 4; j++)
            acc[i][j] = (v8f){0.f, 0.f, 0.f, 0.f, 0.f, 0.f, 0.f, 0.f};

    for (int kk = 0; kk < K; kk += 32) {
        union { v16bf v; v8bf h[2]; } a[4], b[4];
        // A fragment (16x32, 16-bit): lane holds M=lr, K = lh*8..+7 and 16+lh*8..+7
#pragma unroll
        for (int i = 0; i < 4; i++) {
            const unsigned short* p =
                A + (size_t)(m0 + i * 16 + lr) * K + kk + lh * 8;
            a[i].h[0] = *(const v8bf*)(p);
            a[i].h[1] = *(const v8bf*)(p + 16);
        }
        // B fragment (32x16): lane holds N=lr, K = lh*8..+7 and 16+lh*8..+7
#pragma unroll
        for (int j = 0; j < 4; j++) {
            const unsigned short* p =
                Bm + (size_t)(n0 + j * 16 + lr) * K + kk + lh * 8;
            b[j].h[0] = *(const v8bf*)(p);
            b[j].h[1] = *(const v8bf*)(p + 16);
        }
        if (kk + 32 < K) {  // global_prefetch_b8 of next K-slab
            __builtin_prefetch(A + (size_t)(m0 + lr) * K + kk + 32, 0, 0);
            __builtin_prefetch(Bm + (size_t)(n0 + lr) * K + kk + 32, 0, 0);
        }
#pragma unroll
        for (int i = 0; i < 4; i++)
#pragma unroll
            for (int j = 0; j < 4; j++)
                acc[i][j] = __builtin_amdgcn_wmma_f32_16x16x32_bf16(
                    false, a[i].v, false, b[j].v, (short)0, acc[i][j],
                    false, false);
    }

    // Epilogue. C/D layout: VGPR r -> M = r + 8*lh, N = lane&15.
#pragma unroll
    for (int i = 0; i < 4; i++) {
#pragma unroll
        for (int j = 0; j < 4; j++) {
            const int col = n0 + j * 16 + lr;
            const float bs = bias[col];
#pragma unroll
            for (int r = 0; r < 8; r++) {
                const int row = m0 + i * 16 + lh * 8 + r;
                float v = acc[i][j][r] + bs;
                if (EPI == 1) {
                    ((unsigned short*)Out)[(size_t)row * N + col] =
                        f2bf(gelu_erf(v));
                } else {
                    ((float*)Out)[(size_t)row * N + col] = v;
                }
            }
        }
    }
}

// ---------------------------------------------------------------------------
// Top-16 per row of z[SB, SN]; store relu'd values + indices.
// One 256-thread block per row.
// ---------------------------------------------------------------------------
__global__ __launch_bounds__(256) void k_topk(
    const float* __restrict__ z, float* __restrict__ sval, int* __restrict__ sidx) {
    __shared__ float cv[256 * 16];
    __shared__ int   ci[256 * 16];
    __shared__ float rv[256];
    __shared__ int   rp[256];
    const int b = blockIdx.x;
    const int tid = threadIdx.x;
    const float* zr = z + (size_t)b * SN;

    float v[SK]; int id[SK];
#pragma unroll
    for (int t = 0; t < SK; t++) { v[t] = -3.4e38f; id[t] = 0; }
    for (int i = tid; i < SN; i += 256) {
        float x = zr[i];
        if (x > v[SK - 1]) {
            int j = SK - 1;
            while (j > 0 && v[j - 1] < x) { v[j] = v[j - 1]; id[j] = id[j - 1]; j--; }
            v[j] = x; id[j] = i;
        }
    }
#pragma unroll
    for (int t = 0; t < SK; t++) { cv[tid * SK + t] = v[t]; ci[tid * SK + t] = id[t]; }
    __syncthreads();

    for (int r = 0; r < SK; r++) {
        float bm = -3.4e38f; int bp = tid * SK;
#pragma unroll
        for (int t = 0; t < SK; t++) {
            float c = cv[tid * SK + t];
            if (c > bm) { bm = c; bp = tid * SK + t; }
        }
        rv[tid] = bm; rp[tid] = bp;
        __syncthreads();
        for (int s = 128; s > 0; s >>= 1) {
            if (tid < s && rv[tid + s] > rv[tid]) { rv[tid] = rv[tid + s]; rp[tid] = rp[tid + s]; }
            __syncthreads();
        }
        if (tid == 0) {
            int p = rp[0];
            sval[b * SK + r] = fmaxf(cv[p], 0.0f);   // relu(z * mask)
            sidx[b * SK + r] = ci[p];
            cv[p] = -3.4e38f;
        }
        __syncthreads();
    }
}

// ---------------------------------------------------------------------------
// Sparse decode: y[b,d] = gelu( b1[d] + sum_k sval[b,k] * w1T[idx[b,k], d] ) -> bf16
// ---------------------------------------------------------------------------
__global__ __launch_bounds__(256) void k_decode(
    const unsigned short* __restrict__ w1T, const float* __restrict__ sval,
    const int* __restrict__ sidx, const float* __restrict__ b1,
    unsigned short* __restrict__ ybf) {
    __shared__ float sv[SK];
    __shared__ int   si[SK];
    const int b = blockIdx.y;
    const int d = blockIdx.x * 256 + threadIdx.x;
    if (threadIdx.x < SK) {
        sv[threadIdx.x] = sval[b * SK + threadIdx.x];
        si[threadIdx.x] = sidx[b * SK + threadIdx.x];
    }
    __syncthreads();
    if (d >= SD) return;
    float acc = b1[d];
#pragma unroll
    for (int k = 0; k < SK; k++)
        acc += sv[k] * bf2f(w1T[(size_t)si[k] * SD + d]);
    ybf[(size_t)b * SD + d] = f2bf(gelu_erf(acc));
}

// ---------------------------------------------------------------------------
extern "C" void kernel_launch(void* const* d_in, const int* in_sizes, int n_in,
                              void* d_out, int out_size, void* d_ws, size_t ws_size,
                              hipStream_t stream) {
    const float* x   = (const float*)d_in[0];
    const float* ew1 = (const float*)d_in[1];
    const float* eb1 = (const float*)d_in[2];
    const float* ew2 = (const float*)d_in[3];
    const float* eb2 = (const float*)d_in[4];
    const float* dw1 = (const float*)d_in[5];
    const float* db1 = (const float*)d_in[6];
    const float* dw2 = (const float*)d_in[7];
    const float* db2 = (const float*)d_in[8];
    float* out = (float*)d_out;

    char* ws = (char*)d_ws;
    size_t off = 0;
    auto arena = [&](size_t bytes) -> char* {
        char* p = ws + off;
        off += (bytes + 255) & ~(size_t)255;
        return p;
    };
    unsigned short* xs_bf  = (unsigned short*)arena((size_t)SB * SC * 2);  // x - db2
    unsigned short* w1_bf  = (unsigned short*)arena((size_t)SD * SC * 2);  // enc_w1
    unsigned short* w2_bf  = (unsigned short*)arena((size_t)SN * SD * 2);  // enc_w2
    unsigned short* dw2_bf = (unsigned short*)arena((size_t)SC * SD * 2);  // dec_w2
    unsigned short* h_bf   = (unsigned short*)arena((size_t)SB * SD * 2);  // gelu(enc1)
    float*          zf     = (float*)arena((size_t)SB * SN * 4);           // pre-topk
    float*          sval   = (float*)arena((size_t)SB * SK * 4);
    int*            sidx   = (int*)arena((size_t)SB * SK * 4);
    unsigned short* w1T_bf = (unsigned short*)arena((size_t)SN * SD * 2);  // dec_w1^T
    unsigned short* y_bf   = (unsigned short*)arena((size_t)SB * SD * 2);  // gelu(dec1)

    // 1) casts
    {
        size_t n = (size_t)SB * SC;
        k_sub_bf16<<<(unsigned)((n + 255) / 256), 256, 0, stream>>>(x, db2, xs_bf, n);
    }
    {
        size_t n = (size_t)SD * SC;
        k_cvt_bf16<<<(unsigned)((n + 255) / 256), 256, 0, stream>>>(ew1, w1_bf, n);
    }
    {
        size_t n = (size_t)SN * SD;
        k_cvt_bf16<<<(unsigned)((n + 255) / 256), 256, 0, stream>>>(ew2, w2_bf, n);
    }
    {
        size_t n = (size_t)SC * SD;
        k_cvt_bf16<<<(unsigned)((n + 255) / 256), 256, 0, stream>>>(dw2, dw2_bf, n);
    }
    // 2) transpose decoder_w1 for coalesced sparse gather
    k_transpose_bf16<<<dim3(SN / 32, SD / 32), 256, 0, stream>>>(dw1, w1T_bf);

    // 3) GEMM1: h = gelu((x-db2) @ ew1^T + eb1)  -> bf16 [B,D]
    k_gemm_bf16<1><<<dim3(SB / 256, SD / 128), 256, 0, stream>>>(
        xs_bf, w1_bf, eb1, (void*)h_bf, SB, SD, SC);

    // 4) GEMM2: z = h @ ew2^T + eb2  -> f32 [B,N]   (the 825 GFLOP one)
    k_gemm_bf16<0><<<dim3(SB / 256, SN / 128), 256, 0, stream>>>(
        h_bf, w2_bf, eb2, (void*)zf, SB, SN, SD);

    // 5) top-16 + relu per row
    k_topk<<<SB, 256, 0, stream>>>(zf, sval, sidx);

    // 6) sparse decode + gelu -> bf16 [B,D]
    k_decode<<<dim3(SD / 256, SB), 256, 0, stream>>>(w1T_bf, sval, sidx, db1, y_bf);

    // 7) GEMM3: out = y @ dw2^T + db2  -> f32 [B,C]
    k_gemm_bf16<0><<<dim3(SB / 256, SC / 128), 256, 0, stream>>>(
        y_bf, dw2_bf, db2, (void*)out, SB, SC, SD);
}
